// GATModel_42863773614469
// MI455X (gfx1250) — matrix operand order repaired and checked by reading, not compiled, not run on hardware
//
#include <hip/hip_runtime.h>
#include <hip/hip_bf16.h>

#define NN      50000
#define EE      500000
#define IN_DIM  128
#define HEADS   8
#define DH      32
#define HD_DIM  256
#define PERM_D  100
#define BATCH   64
#define CLASSES 10
#define SLOPE   0.2f

typedef __attribute__((ext_vector_type(16))) _Float16 v16h;
typedef __attribute__((ext_vector_type(8)))  _Float16 v8h;
typedef __attribute__((ext_vector_type(8)))  float    v8f;

// ---------------- utility kernels ----------------

__global__ void fill_f32(float* p, float v, int n) {
    int i = blockIdx.x * blockDim.x + threadIdx.x;
    if (i < n) p[i] = v;
}

__global__ void cvt_f32_f16(const float* __restrict__ s, _Float16* __restrict__ d, int n) {
    int i = blockIdx.x * blockDim.x + threadIdx.x;
    if (i < n) d[i] = (_Float16)s[i];
}

// W: [K, HD] f32 row-major  ->  Wt: [HD, K] f16 (each output column contiguous in K)
__global__ void transpose_w_f16(const float* __restrict__ W, _Float16* __restrict__ Wt, int K) {
    int i = blockIdx.x * blockDim.x + threadIdx.x;
    if (i >= K * HD_DIM) return;
    int k = i / HD_DIM, nc = i % HD_DIM;
    Wt[(size_t)nc * K + k] = (_Float16)W[i];
}

// ---------------- WMMA GEMM: C[N,HD] = A16[N,K] @ W (Wt16 = [HD,K]) ----------------
// one wave per (16-row tile x 4 column tiles): A fragment reused across 4 WMMAs/K-step
__global__ void gemm_wmma_f16(const _Float16* __restrict__ A,
                              const _Float16* __restrict__ Bt,
                              float* __restrict__ C, int Nrows, int K) {
    int wave = (blockIdx.x * blockDim.x + threadIdx.x) >> 5;
    int lane = threadIdx.x & 31;
    int tm   = wave >> 2;            // row tile
    int quad = wave & 3;             // which group of 4 column tiles (HD/16 == 16 tiles)
    int m0 = tm * 16, n0 = quad * 64;
    if (m0 >= Nrows) return;

    int half = lane >> 4;            // 0: lanes 0-15, 1: lanes 16-31
    int lr   = lane & 15;

    const _Float16* arow = A + (size_t)(m0 + lr) * K + half * 16;   // A fragment base
    const _Float16* b0 = Bt + (size_t)(n0 +  0 + lr) * K + half * 16;
    const _Float16* b1 = Bt + (size_t)(n0 + 16 + lr) * K + half * 16;
    const _Float16* b2 = Bt + (size_t)(n0 + 32 + lr) * K + half * 16;
    const _Float16* b3 = Bt + (size_t)(n0 + 48 + lr) * K + half * 16;

    v8f c0 = {}, c1 = {}, c2 = {}, c3 = {};
    #pragma unroll 2
    for (int k0 = 0; k0 < K; k0 += 32) {
        // A 16x32 f16: lanes 0-15 -> K k0+0..15, lanes 16-31 -> K k0+16..31 (matches B)
        v8h alo = *(const v8h*)(arow + k0);
        v8h ahi = *(const v8h*)(arow + k0 + 8);
        v16h a = __builtin_shufflevector(alo, ahi, 0,1,2,3,4,5,6,7,8,9,10,11,12,13,14,15);
        // B 32x16 f16: lanes 0-15 -> K k0+0..15, lanes 16-31 -> K k0+16..31, contiguous in Wt
        v16h vb0 = *(const v16h*)(b0 + k0);
        v16h vb1 = *(const v16h*)(b1 + k0);
        v16h vb2 = *(const v16h*)(b2 + k0);
        v16h vb3 = *(const v16h*)(b3 + k0);
        c0 = __builtin_amdgcn_wmma_f32_16x16x32_f16(false, a, false, vb0, (short)0, c0, false, false);
        c1 = __builtin_amdgcn_wmma_f32_16x16x32_f16(false, a, false, vb1, (short)0, c1, false, false);
        c2 = __builtin_amdgcn_wmma_f32_16x16x32_f16(false, a, false, vb2, (short)0, c2, false, false);
        c3 = __builtin_amdgcn_wmma_f32_16x16x32_f16(false, a, false, vb3, (short)0, c3, false, false);
    }
    // C/D layout: VGPR r -> row (half*8 + r), col lr
    float* crow = C + (size_t)(m0 + half * 8) * HD_DIM + n0 + lr;
    #pragma unroll
    for (int r = 0; r < 8; ++r) {
        crow[(size_t)r * HD_DIM +  0] = c0[r];
        crow[(size_t)r * HD_DIM + 16] = c1[r];
        crow[(size_t)r * HD_DIM + 32] = c2[r];
        crow[(size_t)r * HD_DIM + 48] = c3[r];
    }
}

// ---------------- attention coefficients el/er ----------------
__global__ void attn_coeff(const float* __restrict__ z, const float* __restrict__ al,
                           const float* __restrict__ ar, float* __restrict__ el,
                           float* __restrict__ er) {
    int i = blockIdx.x * blockDim.x + threadIdx.x;   // over N*H
    if (i >= NN * HEADS) return;
    int n = i / HEADS, h = i % HEADS;
    const float* zp  = z  + (size_t)n * HD_DIM + h * DH;
    const float* alp = al + h * DH;
    const float* arp = ar + h * DH;
    float sl = 0.f, sr = 0.f;
    #pragma unroll
    for (int d = 0; d < DH; ++d) { float v = zp[d]; sl += v * alp[d]; sr += v * arp[d]; }
    el[i] = sl; er[i] = sr;
}

// sign-split float atomic max (works with -1e30 init)
__device__ inline void atomicMaxF(float* addr, float val) {
    if (val >= 0.f) atomicMax((int*)addr, __float_as_int(val));
    else            atomicMin((unsigned int*)addr, __float_as_uint(val));
}

__global__ void edge_max(const int* __restrict__ src, const int* __restrict__ dst,
                         const float* __restrict__ el, const float* __restrict__ er,
                         float* __restrict__ m) {
    int i = blockIdx.x * blockDim.x + threadIdx.x;   // over E*H
    if (i >= EE * HEADS) return;
    int e = i / HEADS, h = i % HEADS;
    int d = dst[e];
    float x = el[src[e] * HEADS + h] + er[d * HEADS + h];
    x = x > 0.f ? x : SLOPE * x;
    atomicMaxF(&m[d * HEADS + h], x);
}

__global__ void fix_m(float* m, int n) {
    int i = blockIdx.x * blockDim.x + threadIdx.x;
    if (i < n && m[i] < -1e29f) m[i] = 0.f;
}

__global__ void edge_expsum(const int* __restrict__ src, const int* __restrict__ dst,
                            const float* __restrict__ el, const float* __restrict__ er,
                            const float* __restrict__ m, float* __restrict__ ew,
                            float* __restrict__ s) {
    int i = blockIdx.x * blockDim.x + threadIdx.x;   // over E*H
    if (i >= EE * HEADS) return;
    int e = i / HEADS, h = i % HEADS;
    int d = dst[e];
    float x = el[src[e] * HEADS + h] + er[d * HEADS + h];
    x = x > 0.f ? x : SLOPE * x;
    float w = expf(x - m[d * HEADS + h]);
    ew[i] = w;
    atomicAdd(&s[d * HEADS + h], w);
}

// one wave per edge: agg[dst] += alpha[h] * z[src]   (256 floats / edge)
__global__ void edge_aggregate(const int* __restrict__ src, const int* __restrict__ dst,
                               const float* __restrict__ ew, const float* __restrict__ s,
                               const float* __restrict__ z, float* __restrict__ agg) {
    int wave = (blockIdx.x * blockDim.x + threadIdx.x) >> 5;
    int lane = threadIdx.x & 31;
    if (wave >= EE) return;
    int sv = src[wave], dv = dst[wave];
    const float* zrow = z   + (size_t)sv * HD_DIM;
    float*       arow = agg + (size_t)dv * HD_DIM;
    #pragma unroll
    for (int j = 0; j < HEADS; ++j) {
        float alpha = ew[wave * HEADS + j] / s[dv * HEADS + j];
        atomicAdd(&arow[j * DH + lane], alpha * zrow[j * DH + lane]);
    }
}

__global__ void bias_relu_f16(const float* __restrict__ agg, const float* __restrict__ b,
                              _Float16* __restrict__ out16) {
    int i = blockIdx.x * blockDim.x + threadIdx.x;   // over N*HD
    if (i >= NN * HD_DIM) return;
    float v = agg[i] + b[i % HD_DIM];
    out16[i] = (_Float16)(v > 0.f ? v : 0.f);
}

__global__ void pool_nodes(const float* __restrict__ agg, const float* __restrict__ b2,
                           const int* __restrict__ gid, float* __restrict__ pool,
                           float* __restrict__ cnt) {
    int i = blockIdx.x * blockDim.x + threadIdx.x;   // over N*HD
    if (i >= NN * HD_DIM) return;
    int n = i / HD_DIM, c = i % HD_DIM;
    int g = gid[n];
    atomicAdd(&pool[g * HD_DIM + c], agg[i] + b2[c]);
    if (c == 0) atomicAdd(&cnt[g], 1.0f);
}

__global__ void classifier(const float* __restrict__ pool, const float* __restrict__ cnt,
                           const float* __restrict__ perm, const float* __restrict__ Wc,
                           const float* __restrict__ bc, float* __restrict__ out) {
    int i = blockIdx.x * blockDim.x + threadIdx.x;   // over B*C
    if (i >= BATCH * CLASSES) return;
    int b = i / CLASSES, c = i % CLASSES;
    float inv = 1.f / fmaxf(cnt[b], 1.f);
    float acc = bc[c];
    for (int k = 0; k < HD_DIM; ++k)
        acc += pool[b * HD_DIM + k] * inv * Wc[k * CLASSES + c];
    for (int p = 0; p < PERM_D; ++p)
        acc += perm[b * PERM_D + p] * Wc[(HD_DIM + p) * CLASSES + c];
    out[i] = acc;
}

// ---------------- host launcher ----------------

static inline size_t align256(size_t x) { return (x + 255) & ~(size_t)255; }
static inline int blocks(long long n, int bs) { return (int)((n + bs - 1) / bs); }

extern "C" void kernel_launch(void* const* d_in, const int* in_sizes, int n_in,
                              void* d_out, int out_size, void* d_ws, size_t ws_size,
                              hipStream_t stream) {
    const float* h    = (const float*)d_in[0];
    const float* perm = (const float*)d_in[1];
    const float* W1   = (const float*)d_in[2];
    const float* al1  = (const float*)d_in[3];
    const float* ar1  = (const float*)d_in[4];
    const float* b1   = (const float*)d_in[5];
    const float* W2   = (const float*)d_in[6];
    const float* al2  = (const float*)d_in[7];
    const float* ar2  = (const float*)d_in[8];
    const float* b2   = (const float*)d_in[9];
    const float* Wc   = (const float*)d_in[10];
    const float* bc   = (const float*)d_in[11];
    const int*   src  = (const int*)d_in[12];
    const int*   dst  = (const int*)d_in[13];
    const int*   gid  = (const int*)d_in[14];
    float* out = (float*)d_out;

    // workspace carve
    char* ws = (char*)d_ws;
    size_t off = 0;
    auto carve = [&](size_t bytes) { void* p = ws + off; off += align256(bytes); return p; };
    _Float16* Wt1  = (_Float16*)carve((size_t)HD_DIM * IN_DIM * 2);
    _Float16* Wt2  = (_Float16*)carve((size_t)HD_DIM * HD_DIM * 2);
    _Float16* h16  = (_Float16*)carve((size_t)NN * IN_DIM * 2);
    float*    z    = (float*)   carve((size_t)NN * HD_DIM * 4);
    float*    el   = (float*)   carve((size_t)NN * HEADS * 4);
    float*    er   = (float*)   carve((size_t)NN * HEADS * 4);
    float*    mbuf = (float*)   carve((size_t)NN * HEADS * 4);
    float*    sbuf = (float*)   carve((size_t)NN * HEADS * 4);
    float*    ew   = (float*)   carve((size_t)EE * HEADS * 4);
    float*    agg  = (float*)   carve((size_t)NN * HD_DIM * 4);
    _Float16* h116 = (_Float16*)carve((size_t)NN * HD_DIM * 2);
    float*    pool = (float*)   carve((size_t)BATCH * HD_DIM * 4);
    float*    cnt  = (float*)   carve((size_t)BATCH * 4);
    (void)ws_size; (void)n_in; (void)in_sizes; (void)out_size;

    const int BS = 256;
    const long long NH  = (long long)NN * HEADS;
    const long long EH  = (long long)EE * HEADS;
    const long long NHD = (long long)NN * HD_DIM;
    // (N/16) row tiles x 4 column-tile groups, 32 threads per wave
    const int gemm_blocks = blocks((long long)(NN / 16) * 4 * 32, BS);

    // prep: f16 conversions
    cvt_f32_f16<<<blocks((long long)NN * IN_DIM, BS), BS, 0, stream>>>(h, h16, NN * IN_DIM);
    transpose_w_f16<<<blocks((long long)IN_DIM * HD_DIM, BS), BS, 0, stream>>>(W1, Wt1, IN_DIM);
    transpose_w_f16<<<blocks((long long)HD_DIM * HD_DIM, BS), BS, 0, stream>>>(W2, Wt2, HD_DIM);

    // ---------- layer 1 ----------
    gemm_wmma_f16<<<gemm_blocks, BS, 0, stream>>>(h16, Wt1, z, NN, IN_DIM);
    attn_coeff<<<blocks(NH, BS), BS, 0, stream>>>(z, al1, ar1, el, er);
    fill_f32<<<blocks(NH, BS), BS, 0, stream>>>(mbuf, -1e30f, (int)NH);
    fill_f32<<<blocks(NH, BS), BS, 0, stream>>>(sbuf, 0.f, (int)NH);
    edge_max<<<blocks(EH, BS), BS, 0, stream>>>(src, dst, el, er, mbuf);
    fix_m<<<blocks(NH, BS), BS, 0, stream>>>(mbuf, (int)NH);
    edge_expsum<<<blocks(EH, BS), BS, 0, stream>>>(src, dst, el, er, mbuf, ew, sbuf);
    fill_f32<<<blocks(NHD, BS), BS, 0, stream>>>(agg, 0.f, (int)NHD);
    edge_aggregate<<<blocks((long long)EE * 32, BS), BS, 0, stream>>>(src, dst, ew, sbuf, z, agg);
    bias_relu_f16<<<blocks(NHD, BS), BS, 0, stream>>>(agg, b1, h116);

    // ---------- layer 2 ----------
    gemm_wmma_f16<<<gemm_blocks, BS, 0, stream>>>(h116, Wt2, z, NN, HD_DIM);
    attn_coeff<<<blocks(NH, BS), BS, 0, stream>>>(z, al2, ar2, el, er);
    fill_f32<<<blocks(NH, BS), BS, 0, stream>>>(mbuf, -1e30f, (int)NH);
    fill_f32<<<blocks(NH, BS), BS, 0, stream>>>(sbuf, 0.f, (int)NH);
    edge_max<<<blocks(EH, BS), BS, 0, stream>>>(src, dst, el, er, mbuf);
    fix_m<<<blocks(NH, BS), BS, 0, stream>>>(mbuf, (int)NH);
    edge_expsum<<<blocks(EH, BS), BS, 0, stream>>>(src, dst, el, er, mbuf, ew, sbuf);
    fill_f32<<<blocks(NHD, BS), BS, 0, stream>>>(agg, 0.f, (int)NHD);
    edge_aggregate<<<blocks((long long)EE * 32, BS), BS, 0, stream>>>(src, dst, ew, sbuf, z, agg);

    // ---------- readout ----------
    fill_f32<<<blocks((long long)BATCH * HD_DIM, BS), BS, 0, stream>>>(pool, 0.f, BATCH * HD_DIM);
    fill_f32<<<1, BS, 0, stream>>>(cnt, 0.f, BATCH);
    pool_nodes<<<blocks(NHD, BS), BS, 0, stream>>>(agg, b2, gid, pool, cnt);
    classifier<<<blocks((long long)BATCH * CLASSES, BS), BS, 0, stream>>>(pool, cnt, perm, Wc, bc, out);
}
